// Attention_24850680775359
// MI455X (gfx1250) — compile-verified, matrix operand order checked by reading, not run on hardware
//
#include <hip/hip_runtime.h>

#define DIM   768
#define HEADS 12
#define HDIM  64
#define BATCH 2
#define NQ    2048
#define NKV   2048

typedef __attribute__((ext_vector_type(16))) _Float16 v16h;
typedef __attribute__((ext_vector_type(8)))  _Float16 v8h;
typedef __attribute__((ext_vector_type(8)))  float    v8f;

// D = A(16x32 f16) * B(32x16 f16) + C(16x16 f32)
static __device__ __forceinline__ v8f wmma_f16(v16h a, v16h b, v8f c) {
    return __builtin_amdgcn_wmma_f32_16x16x32_f16(false, a, false, b, (short)0, c,
                                                  false, false);
}

// Load a 16x32 A-fragment (or the mirrored B-fragment) from a row-major tile.
// Lane l: row = l&15, K elements {kb..kb+7, kb+16..kb+23}, kb = (l<16 ? 0 : 8).
// Two contiguous 16-byte loads per lane.
static __device__ __forceinline__ v16h frag_f16(const _Float16* tile, int ld, int lane) {
    const int r  = lane & 15;
    const int kb = (lane >> 4) << 3;
    const _Float16* p = tile + (size_t)r * ld + kb;
    v8h lo = *(const v8h*)p;
    v8h hi = *(const v8h*)(p + 16);
    v16h o;
#pragma unroll
    for (int i = 0; i < 8; ++i) { o[i] = lo[i]; o[i + 8] = hi[i]; }
    return o;
}

// ---------------------------------------------------------------------------
// Kernel 1a: convert weights to f16 and transpose (B-operands become N x K).
// ---------------------------------------------------------------------------
__global__ void prep_weights(const float* __restrict__ Wq,
                             const float* __restrict__ Wkv,
                             const float* __restrict__ Wproj,
                             _Float16* __restrict__ WqT, _Float16* __restrict__ WkT,
                             _Float16* __restrict__ WvT, _Float16* __restrict__ WpT) {
    int idx = blockIdx.x * blockDim.x + threadIdx.x;
    if (idx >= DIM * DIM) return;
    int n = idx / DIM;
    int k = idx % DIM;
    WqT[idx] = (_Float16)Wq[(size_t)k * DIM + n];
    WkT[idx] = (_Float16)Wkv[(size_t)k * (2 * DIM) + n];
    WvT[idx] = (_Float16)Wkv[(size_t)k * (2 * DIM) + DIM + n];
    WpT[idx] = (_Float16)Wproj[(size_t)k * DIM + n];
}

// ---------------------------------------------------------------------------
// Kernel 1b: convert activation sources f32 -> f16 (8 elements / thread).
// ---------------------------------------------------------------------------
__global__ void prep_src(const float* __restrict__ qsrc, const float* __restrict__ kvsrc,
                         _Float16* __restrict__ q16, _Float16* __restrict__ kv16) {
    size_t i = ((size_t)blockIdx.x * blockDim.x + threadIdx.x) * 8;
    if (i >= (size_t)BATCH * NQ * DIM) return;
    v8h oq, ok;
#pragma unroll
    for (int j = 0; j < 8; ++j) {
        oq[j] = (_Float16)qsrc[i + j];
        ok[j] = (_Float16)kvsrc[i + j];
    }
    *(v8h*)(q16 + i)  = oq;
    *(v8h*)(kv16 + i) = ok;
}

// ---------------------------------------------------------------------------
// Kernel 2: fused Q/K/V projection. One wave -> one 32x64 output tile
// (two A fragments share the four B fragments: 8 WMMAs per k-step).
// grid = (16 row-tile groups, 12 col groups, 3 {Q,K,V}), block = 256 (8 waves).
// Q,K stored row-major f16 [B*N][768]; V stored transposed: Vt[B][768][NKV].
// ---------------------------------------------------------------------------
__global__ void qkv_kernel(const _Float16* __restrict__ q16, const _Float16* __restrict__ kv16,
                           const _Float16* __restrict__ WqT, const _Float16* __restrict__ WkT,
                           const _Float16* __restrict__ WvT,
                           _Float16* __restrict__ Q, _Float16* __restrict__ K,
                           _Float16* __restrict__ Vt) {
    const int lane = threadIdx.x & 31;
    const int wave = threadIdx.x >> 5;
    const int rt    = blockIdx.x * 8 + wave;   // 32-row tile over B*N = 4096 rows
    const int n0    = blockIdx.y * 64;
    const int which = blockIdx.z;              // 0=Q 1=K 2=V

    const _Float16* src = (which == 0) ? q16 : kv16;
    const _Float16* WT  = (which == 0) ? WqT : (which == 1) ? WkT : WvT;
    const _Float16* Arow0 = src + (size_t)rt * 32 * DIM;
    const _Float16* Arow1 = Arow0 + (size_t)16 * DIM;

    v8f z = {};
    v8f acc[2][4] = {{z, z, z, z}, {z, z, z, z}};
    for (int k = 0; k < DIM; k += 32) {
        v16h a0 = frag_f16(Arow0 + k, DIM, lane);
        v16h a1 = frag_f16(Arow1 + k, DIM, lane);
#pragma unroll
        for (int t = 0; t < 4; ++t) {
            v16h b = frag_f16(WT + (size_t)(n0 + t * 16) * DIM + k, DIM, lane);
            acc[0][t] = wmma_f16(a0, b, acc[0][t]);
            acc[1][t] = wmma_f16(a1, b, acc[1][t]);
        }
    }

    const int off = (lane >> 4) << 3;          // row offset of this half-wave
    if (which == 2) {
        // store V transposed: Vt[b][dim_col][kv_row]; 8 consecutive kv per lane
        const int batch = (rt * 32) >> 11;     // /2048 (32-row tile never crosses)
        const int kvr   = (rt * 32) & 2047;
#pragma unroll
        for (int half = 0; half < 2; ++half) {
#pragma unroll
            for (int t = 0; t < 4; ++t) {
                int d = n0 + t * 16 + (lane & 15);
                v8h pk;
#pragma unroll
                for (int g = 0; g < 8; ++g) pk[g] = (_Float16)acc[half][t][g];
                *(v8h*)(Vt + ((size_t)batch * DIM + d) * NKV + kvr + half * 16 + off) = pk;
            }
        }
    } else {
        _Float16* O = (which == 0) ? Q : K;
#pragma unroll
        for (int half = 0; half < 2; ++half) {
            int m0 = rt * 32 + half * 16 + off;
#pragma unroll
            for (int t = 0; t < 4; ++t) {
                int n = n0 + t * 16 + (lane & 15);
#pragma unroll
                for (int g = 0; g < 8; ++g)
                    O[(size_t)(m0 + g) * DIM + n] = (_Float16)acc[half][t][g];
            }
        }
    }
}

// ---------------------------------------------------------------------------
// Kernel 3: flash attention. One wave owns 16 q-rows x head_dim 64, streams KV.
// grid = (16, HEADS, BATCH), block = 256 (8 waves). Online softmax, scores
// never leave registers/LDS. scale = 1/sqrt(64) = 0.125.
// ---------------------------------------------------------------------------
__global__ void attn_kernel(const _Float16* __restrict__ Qh, const _Float16* __restrict__ Kh,
                            const _Float16* __restrict__ Vt, _Float16* __restrict__ X) {
    __shared__ __align__(16) _Float16 Pls[8][16][32];   // per-wave P staging (8 KB)

    const int lane = threadIdx.x & 31;
    const int wave = threadIdx.x >> 5;
    const int qt = blockIdx.x * 8 + wave;   // 16-row q tile (0..127)
    const int h  = blockIdx.y;
    const int b  = blockIdx.z;

    const _Float16* Qb = Qh + ((size_t)b * NQ + qt * 16) * DIM + h * HDIM;
    const _Float16* Kb = Kh + (size_t)b * NKV * DIM + h * HDIM;
    const _Float16* Vb = Vt + ((size_t)b * DIM + h * HDIM) * NKV;
    _Float16 (&Pl)[16][32] = Pls[wave];

    // Q fragments for d = 0..31 and 32..63 (reused across the whole kv loop)
    v16h aq0 = frag_f16(Qb, DIM, lane);
    v16h aq1 = frag_f16(Qb + 32, DIM, lane);

    v8f z = {};
    v8f acc[4] = {z, z, z, z};                  // 16 x 64 f32 output accumulator
    float mrow[8], lrow[8];
#pragma unroll
    for (int g = 0; g < 8; ++g) { mrow[g] = -3.0e38f; lrow[g] = 0.0f; }

    for (int kv0 = 0; kv0 < NKV; kv0 += 32) {
        // scores for two 16-wide kv tiles: S = Q * K^T (K rows are B-columns)
        v8f sc[2];
#pragma unroll
        for (int t = 0; t < 2; ++t) {
            const _Float16* Krow = Kb + (size_t)(kv0 + t * 16) * DIM;
            v8f c = {};
            c = wmma_f16(aq0, frag_f16(Krow, DIM, lane), c);
            c = wmma_f16(aq1, frag_f16(Krow + 32, DIM, lane), c);
            sc[t] = c;
        }

        // online softmax over these 32 columns; row g lives on the 16 lanes of
        // this half-wave, so xor-shuffles with masks 8..1 reduce a full row.
#pragma unroll
        for (int g = 0; g < 8; ++g) {
            float s0 = sc[0][g] * 0.125f;
            float s1 = sc[1][g] * 0.125f;
            float v = fmaxf(s0, s1);
#pragma unroll
            for (int o = 8; o > 0; o >>= 1) v = fmaxf(v, __shfl_xor(v, o, 32));
            float mnew = fmaxf(mrow[g], v);
            float corr = __expf(mrow[g] - mnew);
            mrow[g] = mnew;
            acc[0][g] *= corr; acc[1][g] *= corr;
            acc[2][g] *= corr; acc[3][g] *= corr;
            float p0 = __expf(s0 - mnew);
            float p1 = __expf(s1 - mnew);
            float ps = p0 + p1;
#pragma unroll
            for (int o = 8; o > 0; o >>= 1) ps += __shfl_xor(ps, o, 32);
            lrow[g] = lrow[g] * corr + ps;
            // stage P (C-layout -> row-major in LDS) for the A-side of P*V
            int m = g + ((lane >> 4) << 3);
            Pl[m][lane & 15]        = (_Float16)p0;
            Pl[m][16 + (lane & 15)] = (_Float16)p1;
        }

        // O += P(16x32) * V(32x64); V stored transposed so B-frags are contiguous
        v16h ap = frag_f16(&Pl[0][0], 32, lane);
#pragma unroll
        for (int t = 0; t < 4; ++t) {
            v16h bv = frag_f16(Vb + (size_t)(t * 16) * NKV + kv0, NKV, lane);
            acc[t] = wmma_f16(ap, bv, acc[t]);
        }
    }

    // normalize and write X (row-major f16) for the output projection
    _Float16* Xb = X + ((size_t)b * NQ + qt * 16) * DIM + h * HDIM;
    const int off = (lane >> 4) << 3;
#pragma unroll
    for (int g = 0; g < 8; ++g) {
        float inv = 1.0f / lrow[g];
        int m = g + off;
#pragma unroll
        for (int t = 0; t < 4; ++t)
            Xb[(size_t)m * DIM + t * 16 + (lane & 15)] = (_Float16)(acc[t][g] * inv);
    }
}

// ---------------------------------------------------------------------------
// Kernel 4: output projection Y = X @ Wproj + bproj (f32 out).
// grid = (16, 12), block = 256 (8 waves), one wave -> 32x64 tile.
// ---------------------------------------------------------------------------
__global__ void proj_kernel(const _Float16* __restrict__ X, const _Float16* __restrict__ WpT,
                            const float* __restrict__ bias, float* __restrict__ out) {
    const int lane = threadIdx.x & 31;
    const int wave = threadIdx.x >> 5;
    const int rt = blockIdx.x * 8 + wave;      // 32-row tile
    const int n0 = blockIdx.y * 64;

    const _Float16* Arow0 = X + (size_t)rt * 32 * DIM;
    const _Float16* Arow1 = Arow0 + (size_t)16 * DIM;
    v8f z = {};
    v8f acc[2][4] = {{z, z, z, z}, {z, z, z, z}};
    for (int k = 0; k < DIM; k += 32) {
        v16h a0 = frag_f16(Arow0 + k, DIM, lane);
        v16h a1 = frag_f16(Arow1 + k, DIM, lane);
#pragma unroll
        for (int t = 0; t < 4; ++t) {
            v16h b = frag_f16(WpT + (size_t)(n0 + t * 16) * DIM + k, DIM, lane);
            acc[0][t] = wmma_f16(a0, b, acc[0][t]);
            acc[1][t] = wmma_f16(a1, b, acc[1][t]);
        }
    }
    const int off = (lane >> 4) << 3;
#pragma unroll
    for (int half = 0; half < 2; ++half) {
        int m0 = rt * 32 + half * 16 + off;
#pragma unroll
        for (int t = 0; t < 4; ++t) {
            int n = n0 + t * 16 + (lane & 15);
            float bv = bias[n];
#pragma unroll
            for (int g = 0; g < 8; ++g)
                out[(size_t)(m0 + g) * DIM + n] = acc[half][t][g] + bv;
        }
    }
}

// ---------------------------------------------------------------------------
extern "C" void kernel_launch(void* const* d_in, const int* in_sizes, int n_in,
                              void* d_out, int out_size, void* d_ws, size_t ws_size,
                              hipStream_t stream) {
    (void)in_sizes; (void)n_in; (void)out_size; (void)ws_size;
    const float* qsrc  = (const float*)d_in[0];
    const float* kvsrc = (const float*)d_in[1];
    const float* Wq    = (const float*)d_in[2];
    const float* Wkv   = (const float*)d_in[3];
    const float* Wproj = (const float*)d_in[4];
    const float* bproj = (const float*)d_in[5];
    float* out = (float*)d_out;

    // workspace carve-up (~30 MB total)
    char* w = (char*)d_ws;
    const size_t WT_BYTES  = (size_t)DIM * DIM * sizeof(_Float16);
    const size_t ACT_BYTES = (size_t)BATCH * NQ * DIM * sizeof(_Float16);
    _Float16* WqT = (_Float16*)w; w += WT_BYTES;
    _Float16* WkT = (_Float16*)w; w += WT_BYTES;
    _Float16* WvT = (_Float16*)w; w += WT_BYTES;
    _Float16* WpT = (_Float16*)w; w += WT_BYTES;
    _Float16* Qh  = (_Float16*)w; w += ACT_BYTES;
    _Float16* Kh  = (_Float16*)w; w += ACT_BYTES;
    _Float16* Vt  = (_Float16*)w; w += ACT_BYTES;   // V transposed: [B][768][NKV]
    _Float16* Xh  = (_Float16*)w; w += ACT_BYTES;

    // f16 source copies, aliased into buffers that are written only later in
    // the pipeline (stream-serialized, rewritten every call -> deterministic):
    //  - q16 aliases Xh   (qkv consumes q16 before attn writes Xh)
    //  - kv16 aliases out (qkv consumes kv16 before proj writes out;
    //    12.6 MB f32 out buffer >= 6.3 MB f16 needed)
    _Float16* q16  = Xh;
    _Float16* kv16 = (_Float16*)out;

    prep_weights<<<dim3((DIM * DIM + 255) / 256), dim3(256), 0, stream>>>(
        Wq, Wkv, Wproj, WqT, WkT, WvT, WpT);
    prep_src<<<dim3(((size_t)BATCH * NQ * DIM / 8 + 255) / 256), dim3(256), 0, stream>>>(
        qsrc, kvsrc, q16, kv16);
    qkv_kernel<<<dim3(16, 12, 3), dim3(256), 0, stream>>>(
        q16, kv16, WqT, WkT, WvT, Qh, Kh, Vt);
    attn_kernel<<<dim3(16, HEADS, BATCH), dim3(256), 0, stream>>>(Qh, Kh, Vt, Xh);
    proj_kernel<<<dim3(16, 12), dim3(256), 0, stream>>>(Xh, WpT, bproj, out);
}
